// SE3TransformerLayer_30580167147936
// MI455X (gfx1250) — matrix operand order repaired and checked by reading, not compiled, not run on hardware
//
#include <hip/hip_runtime.h>
#include <hip/hip_bf16.h>

// SE3-Transformer layer for MI455X (gfx1250, wave32, WMMA).
// All GEMMs run through a bf16-in-LDS / f32-accumulate WMMA core
// (v_wmma_f32_16x16x32_bf16), tiles 64x64x32, 128 threads = 4 waves/block.
// LDS tiles are stored in WMMA *fragment order* (lane fragment = 32 contiguous
// bytes -> 2x ds_load_b128), the K-loop is double-buffered (global float4
// loads for step t+1 issue before the WMMAs of step t), and fp32->bf16
// conversion uses the packed v_cvt_pk_bf16_f32 path via __float22bfloat162_rn.

#define BB  4
#define SS  512
#define HH  768
#define NHH 12
#define DHH 64
#define DFF 3072

typedef __attribute__((ext_vector_type(16))) __bf16 v16bf;
typedef __attribute__((ext_vector_type(8)))  float  v8f;

union FragBF { v16bf v; unsigned short u[16]; };
union FragF  { v8f  v; float          f[8];  };

// packed fp32x2 -> bf16x2 (round-to-nearest-even), returned as one dword
__device__ __forceinline__ unsigned int f2bf2(float lo, float hi) {
  union { __hip_bfloat162 h; unsigned int u; } cv;
  cv.h = __float22bfloat162_rn(make_float2(lo, hi));
  return cv.u;
}

#define BM 64
#define BN 64
#define BK 32

// C[M,N] = A[M,K] * B[K,N]  (TB: B[k][n] = Bp[n*ldb + k]).
// Requirements (all call sites satisfy): M % 64 == 0, K % 32 == 0;
// N unguarded unless GN (only the pos-bias GEMM has ragged N).
//
// Fragment-order LDS layout (ISA 7.12.2, wave32):
//   A element (m, k)  -> sA[(w*32 + m + 16*((k>>3)&1))*16 + ((k&7) + 8*(k>>4))]
//   B element (k, n)  -> sB[((n>>4)*32 + (n&15) + 16*((k>>3)&1))*16 + ((k&7) + 8*(k>>4))]
// so lane L of wave w / n-subtile j reads 16 contiguous halfs at
//   &sA[(w*32+L)*16]  /  &sB[(j*32+L)*16]        (32B, 32B-aligned)
template <bool TB, bool GN, class Epi>
__device__ void gemm_tile(const float* __restrict__ A, long lda,
                          const float* __restrict__ B, long ldb,
                          int M, int N, int K, Epi epi) {
  __shared__ __align__(32) unsigned short sA[2][BM * BK];
  __shared__ __align__(32) unsigned short sB[2][BK * BN];

  const int tid  = threadIdx.x;       // 0..127
  const int lane = tid & 31;
  const int wave = tid >> 5;          // 0..3
  const int m16  = lane & 15;
  const int hh   = lane >> 4;

  const int row0 = blockIdx.y * BM;
  const int col0 = blockIdx.x * BN;

  FragF acc[4];
#pragma unroll
  for (int j = 0; j < 4; ++j)
#pragma unroll
    for (int t = 0; t < 8; ++t) acc[j].f[t] = 0.0f;

  float4 ra[4], rb[4];

  auto load_stage = [&](int k0) {
    // A: 4 x float4 (row r, K-quad kq) — M is a multiple of BM, no guard
#pragma unroll
    for (int i = 0; i < 4; ++i) {
      int g  = i * 128 + tid;          // 0..511
      int r  = g >> 3;                 // 0..63
      int kq = (g & 7) << 2;           // 0,4,...,28
      ra[i] = *(const float4*)&A[(long)(row0 + r) * lda + (k0 + kq)];
    }
    // B: 4 x 4 K-values
#pragma unroll
    for (int i = 0; i < 4; ++i) {
      int g = i * 128 + tid;
      if (TB) {
        int n = g >> 3, kq = (g & 7) << 2;
        int gc = col0 + n;
        if (!GN || gc < N)
          rb[i] = *(const float4*)&B[(long)gc * ldb + (k0 + kq)];
        else
          rb[i] = make_float4(0.f, 0.f, 0.f, 0.f);
      } else {
        int n = g & 63, kq = (g >> 6) << 2;
        int gc = col0 + n;
        if (!GN || gc < N) {
          rb[i].x = B[(long)(k0 + kq + 0) * ldb + gc];
          rb[i].y = B[(long)(k0 + kq + 1) * ldb + gc];
          rb[i].z = B[(long)(k0 + kq + 2) * ldb + gc];
          rb[i].w = B[(long)(k0 + kq + 3) * ldb + gc];
        } else {
          rb[i] = make_float4(0.f, 0.f, 0.f, 0.f);
        }
      }
    }
  };

  auto store_stage = [&](int buf) {
#pragma unroll
    for (int i = 0; i < 4; ++i) {
      int g  = i * 128 + tid;
      int r  = g >> 3;
      int kq = (g & 7) << 2;
      uint2 q = make_uint2(f2bf2(ra[i].x, ra[i].y), f2bf2(ra[i].z, ra[i].w));
      int idx = (((r >> 4) * 32) + (r & 15) + 16 * ((kq >> 3) & 1)) * 16 +
                ((kq & 7) + ((kq >> 4) << 3));
      *(uint2*)&sA[buf][idx] = q;                  // idx % 4 == 0 -> 8B aligned
    }
#pragma unroll
    for (int i = 0; i < 4; ++i) {
      int g = i * 128 + tid;
      int n, kq;
      if (TB) { n = g >> 3; kq = (g & 7) << 2; }
      else    { n = g & 63; kq = (g >> 6) << 2; }
      uint2 q = make_uint2(f2bf2(rb[i].x, rb[i].y), f2bf2(rb[i].z, rb[i].w));
      *(uint2*)&sB[buf][((n >> 4) * 32 + (n & 15) + 16 * ((kq >> 3) & 1)) * 16 + ((kq & 7) + ((kq >> 4) << 3))] = q;
    }
  };

  auto compute = [&](int buf) {
    FragBF a;
    a.v = *(const v16bf*)&sA[buf][(wave * 32 + lane) * 16];
#pragma unroll
    for (int j = 0; j < 4; ++j) {
      FragBF b;
      b.v = *(const v16bf*)&sB[buf][(j * 32 + lane) * 16];
      acc[j].v = __builtin_amdgcn_wmma_f32_16x16x32_bf16(
          false, a.v, false, b.v, (short)0, acc[j].v, false, false);
    }
  };

  // ---- software pipeline: regs + LDS ping-pong, 1 barrier per K-step ----
  const int nsteps = K / BK;
  load_stage(0);
  store_stage(0);
  __syncthreads();
  for (int t = 0; t < nsteps; ++t) {
    int cur = t & 1;
    if (t + 1 < nsteps) load_stage((t + 1) * BK);   // global loads in flight
    compute(cur);                                   // WMMAs from LDS
    if (t + 1 < nsteps) store_stage(1 - cur);       // convert + stage next tile
    __syncthreads();
  }

  // ---- store: C layout — VGPR v holds M = v + 8*(lane>>4), N = lane&15 ----
#pragma unroll
  for (int j = 0; j < 4; ++j)
#pragma unroll
    for (int v = 0; v < 8; ++v) {
      int row = row0 + wave * 16 + v + 8 * hh;
      int col = col0 + j * 16 + m16;
      if (!GN || col < N) epi(row, col, acc[j].f[v]);
    }
}

// ---------------- GEMM kernel instantiations ----------------

__global__ void k_qkv(const float* __restrict__ x, const float* __restrict__ w,
                      const float* __restrict__ bias, float* __restrict__ out) {
  auto epi = [=](int row, int col, float val) {
    out[(long)row * (3 * HH) + col] = val + bias[col];
  };
  gemm_tile<false, false>(x, HH, w, 3 * HH, BB * SS, 3 * HH, HH, epi);
}

__global__ void k_posb(const float* __restrict__ pe, const float* __restrict__ w,
                       const float* __restrict__ bias, float* __restrict__ out) {
  auto epi = [=](int row, int col, float val) {
    out[(long)row * NHH + col] = val + bias[col];
  };
  gemm_tile<false, true>(pe, HH, w, NHH, SS * SS, NHH, HH, epi);
}

__global__ void k_gram(const float* __restrict__ x, const float* __restrict__ norms,
                       float* __restrict__ dist, float* __restrict__ sq) {
  int b = blockIdx.z;
  const float* xb = x + (long)b * SS * HH;
  const float* nb = norms + b * SS;
  float* db = dist + (long)b * SS * SS;
  float* qb = sq + (long)b * SS * SS;
  auto epi = [=](int row, int col, float val) {
    float s = nb[row] + nb[col] - 2.0f * val;    // ||xi-xj||^2 via Gram
    s = fmaxf(s, 0.0f);
    qb[(long)row * SS + col] = s;
    db[(long)row * SS + col] = sqrtf(s);
  };
  gemm_tile<true, false>(xb, HH, xb, HH, SS, SS, HH, epi);
}

__global__ void k_scores(const float* __restrict__ qkv, const float* __restrict__ dist,
                         const float* __restrict__ sq, const float* __restrict__ posb,
                         const float* __restrict__ geo_w, const float* __restrict__ geo_b,
                         const unsigned char* __restrict__ mask,
                         float* __restrict__ scores) {
  int z = blockIdx.z, b = z / NHH, h = z % NHH;
  const float* A  = qkv + (long)b * SS * (3 * HH) + h * DHH;            // q rows
  const float* Bp = qkv + (long)b * SS * (3 * HH) + HH + h * DHH;       // k rows (TB)
  const float w0 = geo_w[h], w1 = geo_w[NHH + h], gb = geo_b[h];
  const float* db = dist + (long)b * SS * SS;
  const float* qb = sq + (long)b * SS * SS;
  const unsigned char* mb = mask + b * SS;
  float* outz = scores + (long)z * SS * SS;
  auto epi = [=](int row, int col, float val) {
    long ij = (long)row * SS + col;
    float s = val * 0.125f + w0 * db[ij] + w1 * qb[ij] + gb + posb[ij * NHH + h];
    s = mb[col] ? s : -1.0e9f;
    outz[ij] = s;
  };
  gemm_tile<true, false>(A, 3 * HH, Bp, 3 * HH, SS, SS, DHH, epi);
}

__global__ void k_ctx(const float* __restrict__ scores, const float* __restrict__ qkv,
                      float* __restrict__ ctx) {
  int z = blockIdx.z, b = z / NHH, h = z % NHH;
  const float* A  = scores + (long)z * SS * SS;                         // attn rows
  const float* Bp = qkv + (long)b * SS * (3 * HH) + 2 * HH + h * DHH;   // v: B[k][n]=Bp[k*ldb+n]
  float* cb = ctx + (long)b * SS * HH + h * DHH;
  auto epi = [=](int row, int col, float val) { cb[(long)row * HH + col] = val; };
  gemm_tile<false, false>(A, SS, Bp, 3 * HH, SS, DHH, SS, epi);
}

__global__ void k_outproj(const float* __restrict__ ctx, const float* __restrict__ w,
                          const float* __restrict__ bias, float* __restrict__ out) {
  auto epi = [=](int row, int col, float val) {
    out[(long)row * HH + col] = val + bias[col];
  };
  gemm_tile<false, false>(ctx, HH, w, HH, BB * SS, HH, HH, epi);
}

__global__ void k_fc1(const float* __restrict__ x1, const float* __restrict__ w,
                      const float* __restrict__ bias, float* __restrict__ out) {
  auto epi = [=](int row, int col, float val) {
    out[(long)row * DFF + col] = fmaxf(val + bias[col], 0.0f);          // relu
  };
  gemm_tile<false, false>(x1, HH, w, DFF, BB * SS, DFF, HH, epi);
}

__global__ void k_fc2(const float* __restrict__ h1, const float* __restrict__ w,
                      const float* __restrict__ bias, float* __restrict__ out) {
  auto epi = [=](int row, int col, float val) {
    out[(long)row * HH + col] = val + bias[col];
  };
  gemm_tile<false, false>(h1, DFF, w, HH, BB * SS, HH, DFF, epi);
}

// ---------------- reductions: norms, softmax, add+layernorm ----------------

__global__ void k_norms(const float* __restrict__ x, float* __restrict__ norms) {
  int row  = blockIdx.x * 4 + (threadIdx.x >> 5);
  int lane = threadIdx.x & 31;
  const float* p = x + (long)row * HH;
  float s = 0.0f;
#pragma unroll
  for (int i = 0; i < HH / 32; ++i) { float v = p[lane + i * 32]; s += v * v; }
  for (int o = 16; o; o >>= 1) s += __shfl_xor(s, o, 32);
  if (lane == 0) norms[row] = s;
}

__global__ void k_softmax(float* __restrict__ scores) {
  long base = (long)blockIdx.x * SS;
  int t = threadIdx.x, wave = t >> 5, lane = t & 31;
  __shared__ float redm[4], reds[4];
  float v[4];
  float mx = -1.0e30f;
#pragma unroll
  for (int i = 0; i < 4; ++i) { v[i] = scores[base + t + i * 128]; mx = fmaxf(mx, v[i]); }
  for (int o = 16; o; o >>= 1) mx = fmaxf(mx, __shfl_xor(mx, o, 32));
  if (lane == 0) redm[wave] = mx;
  __syncthreads();
  mx = fmaxf(fmaxf(redm[0], redm[1]), fmaxf(redm[2], redm[3]));
  float s = 0.0f;
#pragma unroll
  for (int i = 0; i < 4; ++i) { v[i] = __expf(v[i] - mx); s += v[i]; }
  for (int o = 16; o; o >>= 1) s += __shfl_xor(s, o, 32);
  if (lane == 0) reds[wave] = s;
  __syncthreads();
  float inv = 1.0f / (reds[0] + reds[1] + reds[2] + reds[3]);
#pragma unroll
  for (int i = 0; i < 4; ++i) scores[base + t + i * 128] = v[i] * inv;
}

// out = LN(a + b * maskrow) ; one wave per row (24 elems/lane), wave32 shfl reduce
__global__ void k_add_ln(const float* __restrict__ a, const float* __restrict__ bsrc,
                         const unsigned char* __restrict__ mask,
                         const float* __restrict__ g, const float* __restrict__ bb,
                         float* __restrict__ out) {
  int row  = blockIdx.x * 4 + (threadIdx.x >> 5);
  int lane = threadIdx.x & 31;
  const float* pa = a + (long)row * HH;
  const float* pb = bsrc + (long)row * HH;
  float mm = 1.0f;
  if (mask) mm = mask[row] ? 1.0f : 0.0f;
  float vals[HH / 32];
  float s = 0.0f;
#pragma unroll
  for (int i = 0; i < HH / 32; ++i) {
    float v = pa[lane + i * 32] + pb[lane + i * 32] * mm;
    vals[i] = v; s += v;
  }
  for (int o = 16; o; o >>= 1) s += __shfl_xor(s, o, 32);
  float mu = s * (1.0f / HH);
  float vs = 0.0f;
#pragma unroll
  for (int i = 0; i < HH / 32; ++i) { float d = vals[i] - mu; vs += d * d; }
  for (int o = 16; o; o >>= 1) vs += __shfl_xor(vs, o, 32);
  float rs = rsqrtf(vs * (1.0f / HH) + 1e-5f);
  float* po = out + (long)row * HH;
#pragma unroll
  for (int i = 0; i < HH / 32; ++i) {
    int c = lane + i * 32;
    po[c] = (vals[i] - mu) * rs * g[c] + bb[c];
  }
}

// ---------------- driver ----------------

extern "C" void kernel_launch(void* const* d_in, const int* in_sizes, int n_in,
                              void* d_out, int out_size, void* d_ws, size_t ws_size,
                              hipStream_t stream) {
  const float* x     = (const float*)d_in[0];
  const float* pe    = (const float*)d_in[1];
  const unsigned char* mask = (const unsigned char*)d_in[2];
  const float* qkv_w = (const float*)d_in[3];
  const float* qkv_b = (const float*)d_in[4];
  const float* out_w = (const float*)d_in[5];
  const float* out_b = (const float*)d_in[6];
  const float* geo_w = (const float*)d_in[7];
  const float* geo_b = (const float*)d_in[8];
  const float* pos_w = (const float*)d_in[9];
  const float* pos_b = (const float*)d_in[10];
  const float* fc1_w = (const float*)d_in[11];
  const float* fc1_b = (const float*)d_in[12];
  const float* fc2_w = (const float*)d_in[13];
  const float* fc2_b = (const float*)d_in[14];
  const float* ln1_g = (const float*)d_in[15];
  const float* ln1_b = (const float*)d_in[16];
  const float* ln2_g = (const float*)d_in[17];
  const float* ln2_b = (const float*)d_in[18];
  float* out = (float*)d_out;

  float* ws = (float*)d_ws;
  size_t off = 0;
  auto take = [&](size_t n) { float* p = ws + off; off += n; return p; };
  float* qkv    = take((size_t)BB * SS * 3 * HH);      // 18.9 MB
  float* posb   = take((size_t)SS * SS * NHH);         // 12.6 MB
  float* dist   = take((size_t)BB * SS * SS);          //  4 MB
  float* sq     = take((size_t)BB * SS * SS);          //  4 MB
  float* norms  = take((size_t)BB * SS);
  float* scores = take((size_t)BB * NHH * SS * SS);    // 50 MB
  float* ctx    = take((size_t)BB * SS * HH);
  float* ao     = take((size_t)BB * SS * HH);
  float* x1     = take((size_t)BB * SS * HH);
  float* h1     = take((size_t)BB * SS * DFF);         // 25 MB
  float* h2     = take((size_t)BB * SS * HH);

  dim3 blk(128);
  // projections & biases
  k_qkv  <<<dim3((3 * HH) / BN, (BB * SS) / BM), blk, 0, stream>>>(x, qkv_w, qkv_b, qkv);
  k_posb <<<dim3(1, (SS * SS) / BM),             blk, 0, stream>>>(pe, pos_w, pos_b, posb);
  // pairwise distances via Gram matrix
  k_norms<<<(BB * SS) / 4, 128, 0, stream>>>(x, norms);
  k_gram <<<dim3(SS / BN, SS / BM, BB), blk, 0, stream>>>(x, norms, dist, sq);
  // attention
  k_scores <<<dim3(SS / BN, SS / BM, BB * NHH), blk, 0, stream>>>(qkv, dist, sq, posb,
                                                                  geo_w, geo_b, mask, scores);
  k_softmax<<<BB * NHH * SS, 128, 0, stream>>>(scores);
  k_ctx    <<<dim3(1, SS / BM, BB * NHH), blk, 0, stream>>>(scores, qkv, ctx);
  k_outproj<<<dim3(HH / BN, (BB * SS) / BM), blk, 0, stream>>>(ctx, out_w, out_b, ao);
  // residual + LN1
  k_add_ln <<<(BB * SS) / 4, 128, 0, stream>>>(x, ao, (const unsigned char*)nullptr,
                                               ln1_g, ln1_b, x1);
  // FFN
  k_fc1<<<dim3(DFF / BN, (BB * SS) / BM), blk, 0, stream>>>(x1, fc1_w, fc1_b, h1);
  k_fc2<<<dim3(HH / BN, (BB * SS) / BM), blk, 0, stream>>>(h1, fc2_w, fc2_b, h2);
  // mask + residual + LN2 -> output
  k_add_ln<<<(BB * SS) / 4, 128, 0, stream>>>(x1, h2, mask, ln2_g, ln2_b, out);
}
